// RGCNEncoder_14774687498309
// MI455X (gfx1250) — compile-verified, matrix operand order checked by reading, not run on hardware
//
#include <hip/hip_runtime.h>
#include <hip/hip_bf16.h>

// ---------------------------------------------------------------------------
// RGCN encoder (2 layers) for MI455X / gfx1250, wave32 + WMMA bf16.
//
// Layer math:  out = mean_agg(x)[n, r*128+k | x-part] @ [W_r ; root] + bias
// K-big = R*128 + 128 = 1152.  A is [N x 1152] bf16, B is [1152 x 128] bf16.
// ---------------------------------------------------------------------------

#define NNODES 100000
#define NREL   8
#define NBASES 8
#define DIM    128
#define NEDGES 800000
#define KBIG   1152          // NREL*DIM + DIM
#define KTILES 36            // KBIG / 32
#define NTILES 8             // DIM / 16
#define MTILES (NNODES / 16) // 6250
#define APAD   1160          // LDS row stride in halves (580 dwords; 580%64==4
                             // -> 16 rows land on distinct bank groups)

typedef __attribute__((ext_vector_type(16))) __bf16 v16bf;
typedef __attribute__((ext_vector_type(8)))  float  v8f;

union Frag {
    uint4 u[2];
    v16bf v;
};

__device__ __forceinline__ unsigned short f32_to_bf16(float f) {
    unsigned u = __float_as_uint(f);
    unsigned r = u + 0x7FFFu + ((u >> 16) & 1u);   // round to nearest even
    return (unsigned short)(r >> 16);
}
__device__ __forceinline__ float bf16_to_f32(unsigned h) {
    return __uint_as_float(h << 16);
}
__device__ __forceinline__ unsigned pack_bf16x2(float lo, float hi) {
    return (unsigned)f32_to_bf16(lo) | ((unsigned)f32_to_bf16(hi) << 16);
}

// ---------------------------------------------------------------------------
// Zero-fill (16B granules, grid-stride).
// ---------------------------------------------------------------------------
__global__ void zero_kernel(uint4* __restrict__ p, long long n16) {
    long long i = (long long)blockIdx.x * blockDim.x + threadIdx.x;
    long long stride = (long long)gridDim.x * blockDim.x;
    uint4 z = {0u, 0u, 0u, 0u};
    for (; i < n16; i += stride) p[i] = z;
}

// ---------------------------------------------------------------------------
// Build B-pack: W_r = sum_b comp[r,b]*basis[b] (r<8), then root, converted to
// bf16 and laid out per-WMMA-fragment: [kt][nt][lane][16 halves], mirroring the
// documented 16-bit A-matrix layout with N in place of M.
// tid = kt*4096 + nt*512 + lane*16 + e   (36*8*32*16 = 147456 threads)
// ---------------------------------------------------------------------------
__global__ void pack_w_kernel(const float* __restrict__ basis,
                              const float* __restrict__ comp,
                              const float* __restrict__ root,
                              unsigned short* __restrict__ bpack) {
    int tid = blockIdx.x * blockDim.x + threadIdx.x;
    if (tid >= KTILES * NTILES * 32 * 16) return;
    int e    = tid & 15;
    int lane = (tid >> 4) & 31;
    int nt   = (tid >> 9) & 7;
    int kt   = tid >> 12;

    int kb = (lane >> 4) * 8;          // lanes 16-31 hold K+8 block
    int j = e >> 1, s = e & 1;
    int klocal = (j < 4) ? (kb + 2 * j + s) : (kb + 16 + 2 * (j - 4) + s);
    int kbig = kt * 32 + klocal;
    int n    = nt * 16 + (lane & 15);

    float val;
    if (kbig < NREL * DIM) {
        int r = kbig >> 7;
        int k = kbig & 127;
        float acc = 0.f;
#pragma unroll
        for (int b = 0; b < NBASES; ++b)
            acc += comp[r * NBASES + b] * basis[b * DIM * DIM + k * DIM + n];
        val = acc;
    } else {
        val = root[(kbig - NREL * DIM) * DIM + n];
    }
    bpack[tid] = f32_to_bf16(val);
}

// ---------------------------------------------------------------------------
// Scatter: per edge, add bf16(x[src]) into aggbig[dst][rel*128 ..] using the
// CDNA5 packed-bf16 global atomic; one lane also bumps the f32 edge count.
// 64 threads per edge (one per bf16 pair).
// ---------------------------------------------------------------------------
__global__ void scatter_kernel(const int* __restrict__ edge_index,
                               const int* __restrict__ edge_type,
                               const float* __restrict__ x,
                               unsigned short* __restrict__ aggbig,
                               float* __restrict__ cnt) {
    long long tid = (long long)blockIdx.x * blockDim.x + threadIdx.x;
    int pair = (int)(tid & 63);
    int e    = (int)(tid >> 6);
    if (e >= NEDGES) return;
    int src = edge_index[e];
    int dst = edge_index[NEDGES + e];
    int rel = edge_type[e];

    const float2 v = *(const float2*)(x + (size_t)src * DIM + pair * 2);
    unsigned pk = pack_bf16x2(v.x, v.y);
    unsigned long long addr =
        (unsigned long long)(aggbig + (size_t)dst * KBIG + rel * DIM + pair * 2);
    asm volatile("global_atomic_pk_add_bf16 %0, %1, off"
                 :
                 : "v"(addr), "v"(pk)
                 : "memory");
    if (pair == 0) atomicAdd(cnt + (size_t)dst * NREL + rel, 1.0f);
}

// ---------------------------------------------------------------------------
// Finalize: divide each (n, r) segment by max(count,1) in-place (bf16), and
// write the x-part (cols 1024..1151) as bf16.  One thread per dword of a row.
// ---------------------------------------------------------------------------
__global__ void finalize_kernel(unsigned short* __restrict__ aggbig,
                                const float* __restrict__ cnt,
                                const float* __restrict__ x) {
    long long tid = (long long)blockIdx.x * blockDim.x + threadIdx.x;
    int d = (int)(tid % (KBIG / 2));
    int n = (int)(tid / (KBIG / 2));
    if (n >= NNODES) return;
    unsigned* row = (unsigned*)(aggbig + (size_t)n * KBIG);
    if (d < (NREL * DIM) / 2) {
        int r = d >> 6;                       // 64 dwords per relation block
        float c = cnt[(size_t)n * NREL + r];
        float scale = 1.0f / fmaxf(c, 1.0f);
        unsigned v = row[d];
        float lo = bf16_to_f32(v & 0xFFFFu) * scale;
        float hi = bf16_to_f32(v >> 16) * scale;
        row[d] = pack_bf16x2(lo, hi);
    } else {
        int c2 = d - (NREL * DIM) / 2;
        float2 v = *(const float2*)(x + (size_t)n * DIM + c2 * 2);
        row[d] = pack_bf16x2(v.x, v.y);
    }
}

// ---------------------------------------------------------------------------
// GEMM: block = 16 nodes x 128 cols, 8 waves; wave w owns a 16x16 f32 tile at
// column-tile w.  The 16x1152 bf16 A tile (36 KB) is staged into LDS once per
// block with CDNA5 async global->LDS copies (ASYNCcnt), then all 8 waves read
// their A fragments from LDS (ds_load_b128, bank-padded rows).  B fragments
// come pre-packed from global (288 KB total, L2-resident broadcast).
// K = 1152 in 36 steps of v_wmma_f32_16x16x32_bf16.
// ---------------------------------------------------------------------------
__global__ void __launch_bounds__(256)
gemm_kernel(const unsigned short* __restrict__ aggbig,
            const unsigned short* __restrict__ bpack,
            const float* __restrict__ bias,
            float* __restrict__ out, int relu) {
    __shared__ unsigned short atile[16 * APAD];   // 37,120 B

    int tile = blockIdx.x;
    int t    = threadIdx.x;

    // ---- async-stage A tile: 2304 x 16B chunks, 9 per thread -------------
    const unsigned short* abase = aggbig + (size_t)tile * 16 * KBIG;
#pragma unroll
    for (int i = 0; i < 9; ++i) {
        int c   = t + i * 256;        // chunk id in [0, 2304)
        int row = c / 144;            // 144 chunks of 16B per 2304B row
        int off = (c % 144) * 8;      // offset in halves
        unsigned long long gaddr =
            (unsigned long long)(abase + (size_t)row * KBIG + off);
        unsigned laddr = (unsigned)(size_t)(atile + row * APAD + off);
        asm volatile("global_load_async_to_lds_b128 %0, %1, off"
                     :
                     : "v"(laddr), "v"(gaddr)
                     : "memory");
    }
    asm volatile("s_wait_asynccnt 0x0" ::: "memory");
    __syncthreads();

    // ---- WMMA main loop ---------------------------------------------------
    int w    = t >> 5;
    int lane = t & 31;
    int kb   = (lane >> 4) * 8;
    const unsigned short* alds = atile + (lane & 15) * APAD + kb;
    const unsigned short* bptr = bpack + ((size_t)w * 32 + lane) * 16;

    v8f acc = {};
#pragma unroll 4
    for (int kt = 0; kt < KTILES; ++kt) {
        Frag a, b;
        a.u[0] = *(const uint4*)(alds + kt * 32);
        a.u[1] = *(const uint4*)(alds + kt * 32 + 16);
        b.u[0] = *(const uint4*)(bptr);
        b.u[1] = *(const uint4*)(bptr + 8);
        bptr += NTILES * 32 * 16;   // next kt
        acc = __builtin_amdgcn_wmma_f32_16x16x32_bf16(
            false, a.v, false, b.v, (short)0, acc, false, false);
    }

    int col = w * 16 + (lane & 15);
    float bv = bias[col];
    int mbase = tile * 16 + (lane >> 4) * 8;   // C layout: vgpr i -> row mbase+i
#pragma unroll
    for (int i = 0; i < 8; ++i) {
        float v = acc[i] + bv;
        if (relu) v = fmaxf(v, 0.f);
        out[(size_t)(mbase + i) * DIM + col] = v;
    }
}

// ---------------------------------------------------------------------------
// Host orchestration.
// Workspace layout (bytes):
//   aggbig bf16 [N x 1152]          : 230,400,000
//   cnt f32     [N x R]             :   3,200,000
//   x1 f32      [N x 128]           :  51,200,000
//   bpack1 bf16 [36*8*32*16]        :     294,912
//   bpack2 bf16 [36*8*32*16]        :     294,912
// ---------------------------------------------------------------------------
extern "C" void kernel_launch(void* const* d_in, const int* in_sizes, int n_in,
                              void* d_out, int out_size, void* d_ws, size_t ws_size,
                              hipStream_t stream) {
    const int*   edge_index = (const int*)d_in[0];
    const int*   edge_type  = (const int*)d_in[1];
    const float* emb        = (const float*)d_in[2];
    const float* basis1     = (const float*)d_in[3];
    const float* comp1      = (const float*)d_in[4];
    const float* root1      = (const float*)d_in[5];
    const float* bias1      = (const float*)d_in[6];
    const float* basis2     = (const float*)d_in[7];
    const float* comp2      = (const float*)d_in[8];
    const float* root2      = (const float*)d_in[9];
    const float* bias2      = (const float*)d_in[10];
    float* out = (float*)d_out;

    char* ws = (char*)d_ws;
    unsigned short* aggbig = (unsigned short*)(ws);
    size_t off = (size_t)NNODES * KBIG * 2;                 // 230,400,000
    float* cnt = (float*)(ws + off);
    off += (size_t)NNODES * NREL * 4;                        // +3,200,000
    float* x1 = (float*)(ws + off);
    off += (size_t)NNODES * DIM * 4;                         // +51,200,000
    unsigned short* bpack1 = (unsigned short*)(ws + off);
    off += (size_t)KTILES * NTILES * 32 * 16 * 2;            // +294,912
    unsigned short* bpack2 = (unsigned short*)(ws + off);

    const long long agg16 = ((long long)NNODES * KBIG * 2) / 16;  // uint4 count
    const long long cnt16 = ((long long)NNODES * NREL * 4) / 16;

    // Pack both layers' weights (basis combine + root, bf16 fragment layout).
    {
        int threads = KTILES * NTILES * 32 * 16;
        int blocks = (threads + 255) / 256;
        pack_w_kernel<<<blocks, 256, 0, stream>>>(basis1, comp1, root1, bpack1);
        pack_w_kernel<<<blocks, 256, 0, stream>>>(basis2, comp2, root2, bpack2);
    }

    const long long scatter_threads  = (long long)NEDGES * 64;
    const int       scatter_blocks   = (int)((scatter_threads + 255) / 256);
    const long long finalize_threads = (long long)NNODES * (KBIG / 2);
    const int       finalize_blocks  = (int)((finalize_threads + 255) / 256);

    // ---------------- Layer 1 ----------------
    zero_kernel<<<4096, 256, 0, stream>>>((uint4*)aggbig, agg16);
    zero_kernel<<<256, 256, 0, stream>>>((uint4*)cnt, cnt16);
    scatter_kernel<<<scatter_blocks, 256, 0, stream>>>(edge_index, edge_type,
                                                       emb, aggbig, cnt);
    finalize_kernel<<<finalize_blocks, 256, 0, stream>>>(aggbig, cnt, emb);
    gemm_kernel<<<MTILES, 256, 0, stream>>>(aggbig, bpack1, bias1, x1, 1);

    // ---------------- Layer 2 ----------------
    zero_kernel<<<4096, 256, 0, stream>>>((uint4*)aggbig, agg16);
    zero_kernel<<<256, 256, 0, stream>>>((uint4*)cnt, cnt16);
    scatter_kernel<<<scatter_blocks, 256, 0, stream>>>(edge_index, edge_type,
                                                       x1, aggbig, cnt);
    finalize_kernel<<<finalize_blocks, 256, 0, stream>>>(aggbig, cnt, x1);
    gemm_kernel<<<MTILES, 256, 0, stream>>>(aggbig, bpack2, bias2, out, 0);
}
